// LmmseBaselineModel_82626580840902
// MI455X (gfx1250) — compile-verified, matrix operand order checked by reading, not run on hardware
//
#include <hip/hip_runtime.h>
#include <hip/hip_bf16.h>
#include <math.h>

// ---------------- problem constants (from reference) ----------------
#define BATCH    1000
#define NUE_     4
#define NBS_     4
#define K_       500
#define M_       500
#define N_       1000
#define BPS_     4
#define NSYM_    250
#define NITER_   5
#define NEDGE    2000          // K*3 + M
#define NCOL     4000          // batch * NUE
#define NSYMTOT  250000        // batch * NSYM
#define ROWS_    4000          // batch * NUE  (GEMM rows)
#define KPAD     512           // K padded for WMMA (16 chunks of 32)
#define PAR_STRIDE 512         // padded parity columns (500 -> 512)

typedef __attribute__((ext_vector_type(16))) _Float16 v16h;
typedef __attribute__((ext_vector_type(8)))  _Float16 v8h;   // 16 bytes -> global_load_b128
typedef __attribute__((ext_vector_type(8)))  float    v8f;

// ---------------- small complex helpers ----------------
struct cpx { float re, im; };
__device__ __forceinline__ cpx cmk(float r, float i) { cpx c; c.re = r; c.im = i; return c; }
__device__ __forceinline__ cpx cadd(cpx a, cpx b) { return cmk(a.re + b.re, a.im + b.im); }
__device__ __forceinline__ cpx csub(cpx a, cpx b) { return cmk(a.re - b.re, a.im - b.im); }
__device__ __forceinline__ cpx cmul(cpx a, cpx b) { return cmk(a.re*b.re - a.im*b.im, a.re*b.im + a.im*b.re); }
__device__ __forceinline__ cpx cmulcj(cpx a, cpx b) { // a * conj(b)
    return cmk(a.re*b.re + a.im*b.im, a.im*b.re - a.re*b.im);
}
__device__ __forceinline__ cpx cconj(cpx a) { return cmk(a.re, -a.im); }
__device__ __forceinline__ cpx cscale(cpx a, float s) { return cmk(a.re*s, a.im*s); }
__device__ __forceinline__ float cabs2(cpx a) { return a.re*a.re + a.im*a.im; }

// 16-QAM point for label p (matches reference POINTS ordering)
__device__ __forceinline__ void qam_point(int p, float& re, float& im) {
    const float inv_sqrt10 = 0.31622776601683794f;
    float b0 = (float)((p >> 3) & 1);
    float b1 = (float)((p >> 2) & 1);
    float b2 = (float)((p >> 1) & 1);
    float b3 = (float)(p & 1);
    re = (1.f - 2.f*b0) * (2.f - (1.f - 2.f*b2)) * inv_sqrt10;
    im = (1.f - 2.f*b1) * (2.f - (1.f - 2.f*b3)) * inv_sqrt10;
}

// ---------------- kernel 0: bf output copy ----------------
__global__ void copy_bf_kernel(const int* __restrict__ b, float* __restrict__ out, int n) {
    int i = blockIdx.x * blockDim.x + threadIdx.x;
    if (i < n) out[i] = (float)b[i];
}

// ---------------- kernel 1: zero c2v ----------------
__global__ void zero_f32_kernel(float* __restrict__ p, int n) {
    int i = blockIdx.x * blockDim.x + threadIdx.x;
    if (i < n) p[i] = 0.f;
}

// ---------------- kernel 2a: convert A = f16(b), zero-padded to 4000x512 ----------------
__global__ void convert_a_kernel(const int* __restrict__ b, _Float16* __restrict__ Ah) {
    int i = blockIdx.x * blockDim.x + threadIdx.x;   // row*KPAD + k
    if (i >= ROWS_ * KPAD) return;
    int row = i >> 9;          // /512
    int k   = i & (KPAD - 1);
    Ah[i] = (k < K_) ? (_Float16)(float)b[row * K_ + k] : (_Float16)0.f;
}

// ---------------- kernel 2b: convert B = f16(P)^T, zero-padded to 512x512 ------------
// Bt[n][k] = P[k][n]  -> lane fragment loads become contiguous along k
__global__ void convert_b_kernel(const int* __restrict__ P, _Float16* __restrict__ Bt) {
    int i = blockIdx.x * blockDim.x + threadIdx.x;   // n*KPAD + k
    if (i >= KPAD * KPAD) return;
    int n = i >> 9;
    int k = i & (KPAD - 1);
    Bt[i] = (k < K_ && n < M_) ? (_Float16)(float)P[k * M_ + n] : (_Float16)0.f;
}

// ---------------- kernel 3: parity GEMM via WMMA ----------------
// S = bf(4000x500) @ P(500x500); parity = round(S) mod 2.
// Exact in f16 x f16 -> f32 (integer sums <= 500 < 2048).
// One wave per 16x16 tile. Fragment layout (ISA 7.12.2, f16 16x32): each lane's
// 16 elements are two contiguous runs of 8 halfs -> two b128 loads per operand.
__global__ __launch_bounds__(256)
void parity_gemm_kernel(const _Float16* __restrict__ Ah, const _Float16* __restrict__ Bt,
                        unsigned char* __restrict__ par) {
    const int NT = 32;                      // col tiles (ceil(500/16))
    int wave = threadIdx.x >> 5;
    int lane = threadIdx.x & 31;
    int tile = blockIdx.x * 8 + wave;       // 250*32 = 8000 tiles, 1000 blocks * 8 waves
    int nt = tile % NT;
    int mt = tile / NT;                     // 0..249
    int half_id = lane >> 4;                // lane group (0: lanes 0-15, 1: lanes 16-31)
    int lme = lane & 15;

    const _Float16* arow = Ah + (size_t)(mt * 16 + lme) * KPAD + half_id * 8;
    const _Float16* brow = Bt + (size_t)(nt * 16 + lme) * KPAD + half_id * 8;

    v8f acc = (v8f)0.f;
    #pragma unroll 4
    for (int kc = 0; kc < 16; ++kc) {
        int kbase = kc * 32;
        v8h a0 = *(const v8h*)(arow + kbase);        // k = kbase+half*8 .. +7
        v8h a1 = *(const v8h*)(arow + kbase + 16);   // k = kbase+16+half*8 .. +7
        v8h b0 = *(const v8h*)(brow + kbase);
        v8h b1 = *(const v8h*)(brow + kbase + 16);
        v16h afrag = __builtin_shufflevector(a0, a1, 0,1,2,3,4,5,6,7,8,9,10,11,12,13,14,15);
        v16h bfrag = __builtin_shufflevector(b0, b1, 0,1,2,3,4,5,6,7,8,9,10,11,12,13,14,15);
        acc = __builtin_amdgcn_wmma_f32_16x16x32_f16(
            false, afrag, false, bfrag, (short)0, acc, false, false);
    }
    // C/D layout: lane holds column n = lane%16; VGPR r -> row m = r + 8*(lane/16)
    #pragma unroll
    for (int r = 0; r < 8; ++r) {
        int m = mt * 16 + r + 8 * half_id;
        int n = nt * 16 + lme;
        if (n < M_) {
            int s = (int)(acc[r] + 0.5f);   // exact integer sum
            par[m * PAR_STRIDE + n] = (unsigned char)(s & 1);
        }
    }
}

// ---------------- kernel 4: pack 4-bit symbol labels ----------------
__global__ void pack_symbols_kernel(const int* __restrict__ b,
                                    const unsigned char* __restrict__ par,
                                    unsigned char* __restrict__ idxs) {
    int gid = blockIdx.x * blockDim.x + threadIdx.x;   // r*NSYM + s
    if (gid >= ROWS_ * NSYM_) return;
    int r = gid / NSYM_;
    int s = gid - r * NSYM_;
    int p = 0;
    #pragma unroll
    for (int j = 0; j < 4; ++j) {
        int n = s * 4 + j;
        int bit = (n < K_) ? (b[r * K_ + n] & 1) : (int)par[r * PAR_STRIDE + (n - K_)];
        p = (p << 1) | bit;
    }
    idxs[gid] = (unsigned char)p;
}

// ---------------- Cholesky solve (4x4 Hermitian PD) ----------------
__device__ __forceinline__ void chol_solve(const cpx L[4][4], const float Ld[4],
                                           const cpx rhs[4], cpx out[4]) {
    cpx u[4];
    #pragma unroll
    for (int i = 0; i < 4; ++i) {
        cpx v = rhs[i];
        #pragma unroll
        for (int k = 0; k < 4; ++k)
            if (k < i) v = csub(v, cmul(L[i][k], u[k]));
        u[i] = cscale(v, 1.f / Ld[i]);
    }
    #pragma unroll
    for (int i = 3; i >= 0; --i) {
        cpx v = u[i];
        #pragma unroll
        for (int k = 0; k < 4; ++k)
            if (k > i) v = csub(v, cmul(cconj(L[k][i]), out[k]));
        out[i] = cscale(v, 1.f / Ld[i]);
    }
}

// ---------------- kernel 5: channel + LMMSE + max-log LLR ----------------
__global__ __launch_bounds__(256)
void lmmse_kernel(const float* __restrict__ h_re, const float* __restrict__ h_im,
                  const float* __restrict__ n_re, const float* __restrict__ n_im,
                  const unsigned char* __restrict__ idxs,
                  const float* __restrict__ ebno,
                  float* __restrict__ Lch) {
    int s = blockIdx.x * blockDim.x + threadIdx.x;
    if (s >= NSYMTOT) return;
    int bi  = s / NSYM_;
    int sym = s - bi * NSYM_;

    float no = 1.f / (powf(10.f, ebno[0] * 0.1f) * (float)(BPS_) * 0.5f); // BPS*RATE=2
    const float inv_sqrt2 = 0.7071067811865476f;

    // h = (h_re + i h_im)/sqrt(2)
    cpx h[4][4];
    #pragma unroll
    for (int i = 0; i < 4; ++i)
        #pragma unroll
        for (int j = 0; j < 4; ++j) {
            int o = s * 16 + i * 4 + j;
            h[i][j] = cmk(h_re[o] * inv_sqrt2, h_im[o] * inv_sqrt2);
        }

    // transmitted symbols for this resource element
    cpx x[4];
    #pragma unroll
    for (int ue = 0; ue < 4; ++ue) {
        int p = (int)idxs[(bi * NUE_ + ue) * NSYM_ + sym];
        float pr, pi; qam_point(p, pr, pi);
        x[ue] = cmk(pr, pi);
    }

    // y = h x + w,   w = noise * sqrt(no/2)
    float wsc = sqrtf(no * 0.5f);
    cpx y[4];
    #pragma unroll
    for (int i = 0; i < 4; ++i) {
        cpx acc = cmk(n_re[s * 4 + i] * wsc, n_im[s * 4 + i] * wsc);
        #pragma unroll
        for (int j = 0; j < 4; ++j) acc = cadd(acc, cmul(h[i][j], x[j]));
        y[i] = acc;
    }

    // A = h h^H + no I   (store lower triangle)
    cpx A[4][4];
    #pragma unroll
    for (int i = 0; i < 4; ++i)
        #pragma unroll
        for (int j = 0; j < 4; ++j) {
            if (j <= i) {
                cpx acc = cmk((i == j) ? no : 0.f, 0.f);
                #pragma unroll
                for (int k = 0; k < 4; ++k) acc = cadd(acc, cmulcj(h[i][k], h[j][k]));
                A[i][j] = acc;
            } else A[i][j] = cmk(0.f, 0.f);
        }

    // Cholesky A = L L^H
    cpx L[4][4];
    float Ld[4];
    #pragma unroll
    for (int j = 0; j < 4; ++j) {
        float d = A[j][j].re;
        #pragma unroll
        for (int k = 0; k < 4; ++k)
            if (k < j) d -= cabs2(L[j][k]);
        d = sqrtf(fmaxf(d, 1e-20f));
        Ld[j] = d;
        float invd = 1.f / d;
        #pragma unroll
        for (int i = 0; i < 4; ++i) {
            if (i > j) {
                cpx v = A[i][j];
                #pragma unroll
                for (int k = 0; k < 4; ++k)
                    if (k < j) v = csub(v, cmulcj(L[i][k], L[j][k]));
                L[i][j] = cscale(v, invd);
            } else L[i][j] = cmk(0.f, 0.f);
        }
    }

    // z = A^{-1} y
    cpx z[4];
    chol_solve(L, Ld, y, z);

    int colbase = bi * NUE_;
    #pragma unroll
    for (int ue = 0; ue < 4; ++ue) {
        cpx col[4], tc[4];
        #pragma unroll
        for (int i = 0; i < 4; ++i) col[i] = h[i][ue];
        chol_solve(L, Ld, col, tc);
        float d = 0.f;
        cpx xr = cmk(0.f, 0.f);
        #pragma unroll
        for (int i = 0; i < 4; ++i) {
            d  += col[i].re * tc[i].re + col[i].im * tc[i].im;  // Re(conj(col)·tc)
            xr  = cadd(xr, cmulcj(z[i], col[i]));               // conj(col)·z
        }
        float invd  = 1.f / d;
        float xhr   = xr.re * invd, xhi = xr.im * invd;
        float noeff = fmaxf(invd - 1.f, 1e-12f);
        float inv_noeff = 1.f / noeff;

        float m0[4] = {-1e30f, -1e30f, -1e30f, -1e30f};
        float m1[4] = {-1e30f, -1e30f, -1e30f, -1e30f};
        #pragma unroll
        for (int p = 0; p < 16; ++p) {
            float pr, pi; qam_point(p, pr, pi);
            float dr = xhr - pr, di = xhi - pi;
            float met = -(dr * dr + di * di) * inv_noeff;
            #pragma unroll
            for (int j = 0; j < 4; ++j) {
                if ((p >> (3 - j)) & 1) m1[j] = fmaxf(m1[j], met);
                else                    m0[j] = fmaxf(m0[j], met);
            }
        }
        int ci = colbase + ue;
        #pragma unroll
        for (int j = 0; j < 4; ++j)
            Lch[(size_t)(sym * 4 + j) * NCOL + ci] = m0[j] - m1[j];
    }
}

// ---------------- kernel 6: LDPC BP (thread = codeword column) ----------------
__global__ __launch_bounds__(256)
void ldpc_kernel(const int* __restrict__ cn_idx, const int* __restrict__ vn_idx,
                 const float* __restrict__ Lch,
                 float* __restrict__ c2v, float* __restrict__ tbuf,
                 float* __restrict__ vtot, float* __restrict__ prod,
                 float* __restrict__ out_bhat) {
    __shared__ int svn[NEDGE];
    __shared__ int scn[NEDGE];
    for (int e = threadIdx.x; e < NEDGE; e += blockDim.x) {
        svn[e] = vn_idx[e];
        scn[e] = cn_idx[e];
    }
    __syncthreads();

    int col = blockIdx.x * blockDim.x + threadIdx.x;
    if (col >= NCOL) return;
    const int C = NCOL;

    for (int it = 0; it < NITER_; ++it) {
        // vtot = Lch + scatter_add(c2v)
        for (int n = 0; n < N_; ++n) vtot[(size_t)n * C + col] = Lch[(size_t)n * C + col];
        for (int e = 0; e < NEDGE; ++e)
            vtot[(size_t)svn[e] * C + col] += c2v[(size_t)e * C + col];
        // t = clamp(tanh((vtot[vn]-c2v)/2)); prod = segment_prod(t, cn)
        for (int m = 0; m < M_; ++m) prod[(size_t)m * C + col] = 1.f;
        for (int e = 0; e < NEDGE; ++e) {
            float mvc = vtot[(size_t)svn[e] * C + col] - c2v[(size_t)e * C + col];
            float hv = fminf(fmaxf(0.5f * mvc, -9.9f), 9.9f);
            float tv = tanhf(hv);
            tv = (tv >= 0.f) ? fmaxf(tv, 1e-7f) : fminf(tv, -1e-7f);
            tbuf[(size_t)e * C + col] = tv;
            prod[(size_t)scn[e] * C + col] *= tv;
        }
        // c2v = 2*atanh(clip(prod[cn]/t))
        for (int e = 0; e < NEDGE; ++e) {
            float r = prod[(size_t)scn[e] * C + col] / tbuf[(size_t)e * C + col];
            r = fminf(fmaxf(r, -0.999999f), 0.999999f);
            c2v[(size_t)e * C + col] = 2.f * atanhf(r);
        }
    }
    // final vtot and hard decisions on systematic bits
    for (int n = 0; n < N_; ++n) vtot[(size_t)n * C + col] = Lch[(size_t)n * C + col];
    for (int e = 0; e < NEDGE; ++e)
        vtot[(size_t)svn[e] * C + col] += c2v[(size_t)e * C + col];
    for (int n = 0; n < K_; ++n)
        out_bhat[(size_t)col * K_ + n] = (vtot[(size_t)n * C + col] < 0.f) ? 1.f : 0.f;
}

// ---------------- launch ----------------
extern "C" void kernel_launch(void* const* d_in, const int* in_sizes, int n_in,
                              void* d_out, int out_size, void* d_ws, size_t ws_size,
                              hipStream_t stream) {
    (void)in_sizes; (void)n_in; (void)out_size; (void)ws_size;
    const float* ebno = (const float*)d_in[1];
    const int*   b    = (const int*)d_in[2];
    const int*   P    = (const int*)d_in[3];
    const int*   cn   = (const int*)d_in[4];
    const int*   vn   = (const int*)d_in[5];
    const float* h_re = (const float*)d_in[6];
    const float* h_im = (const float*)d_in[7];
    const float* n_re = (const float*)d_in[8];
    const float* n_im = (const float*)d_in[9];

    float* out_bf   = (float*)d_out;
    float* out_bhat = (float*)d_out + (size_t)NCOL * K_;

    // workspace layout (~112 MB; working set fits MI455X's 192 MB L2)
    char* w = (char*)d_ws;
    unsigned char* par  = (unsigned char*)w;  w += (size_t)ROWS_ * PAR_STRIDE;   // 2.048 MB
    unsigned char* idxs = (unsigned char*)w;  w += (size_t)ROWS_ * NSYM_;        // 1.0 MB
    w += 48;  // pad to 16B alignment for f16 staging (3,048,000 + 48 = 3,048,048? keep 16B)
    w = (char*)(((size_t)w + 15) & ~(size_t)15);
    _Float16* Ah = (_Float16*)w;  w += (size_t)ROWS_ * KPAD * 2;                 // 4.0 MB
    _Float16* Bt = (_Float16*)w;  w += (size_t)KPAD * KPAD * 2;                  // 0.5 MB
    float* Lch  = (float*)w;  w += (size_t)N_    * NCOL * 4;                     // 16 MB
    float* c2v  = (float*)w;  w += (size_t)NEDGE * NCOL * 4;                     // 32 MB
    float* tbuf = (float*)w;  w += (size_t)NEDGE * NCOL * 4;                     // 32 MB
    float* vtot = (float*)w;  w += (size_t)N_    * NCOL * 4;                     // 16 MB
    float* prod = (float*)w;  w += (size_t)M_    * NCOL * 4;                     // 8 MB

    // 0) bf output = float(b)
    {
        int n = NCOL * K_;
        copy_bf_kernel<<<(n + 255) / 256, 256, 0, stream>>>(b, out_bf, n);
    }
    // 1) c2v = 0 (must re-init every launch: ws is not re-poisoned)
    {
        int n = NEDGE * NCOL;
        zero_f32_kernel<<<(n + 255) / 256, 256, 0, stream>>>(c2v, n);
    }
    // 2) stage f16 operands (zero-padded; B transposed for contiguous fragments)
    {
        int n = ROWS_ * KPAD;
        convert_a_kernel<<<(n + 255) / 256, 256, 0, stream>>>(b, Ah);
        int m = KPAD * KPAD;
        convert_b_kernel<<<(m + 255) / 256, 256, 0, stream>>>(P, Bt);
    }
    // 3) parity GEMM (WMMA): 250 x 32 tiles, 8 waves/block
    parity_gemm_kernel<<<1000, 256, 0, stream>>>(Ah, Bt, par);
    // 4) pack symbol labels
    {
        int n = ROWS_ * NSYM_;
        pack_symbols_kernel<<<(n + 255) / 256, 256, 0, stream>>>(b, par, idxs);
    }
    // 5) channel + LMMSE + LLR
    lmmse_kernel<<<(NSYMTOT + 255) / 256, 256, 0, stream>>>(h_re, h_im, n_re, n_im,
                                                            idxs, ebno, Lch);
    // 6) LDPC BP + hard decisions
    ldpc_kernel<<<(NCOL + 255) / 256, 256, 0, stream>>>(cn, vn, Lch, c2v, tbuf,
                                                        vtot, prod, out_bhat);
}
